// S4Layer_39556648796558
// MI455X (gfx1250) — matrix operand order repaired and checked by reading, not compiled, not run on hardware
//
#include <hip/hip_runtime.h>
#include <hip/hip_bf16.h>

// ---------------------------------------------------------------------------
// S4 layer forward for MI455X (gfx1250).  H=256, N=64, L=4096, B=8.
// FFT-convolution pipeline (bandwidth-bound, ~170MB traffic @ 23.3 TB/s);
// WMMA (v_wmma_f32_16x16x32_f16) used for the 12 repeated squarings that
// compute Abar^4096 per channel.
// ---------------------------------------------------------------------------

#define Hc 256
#define Nc 64
#define Lc 4096
#define Bc 8
#define L2 8192

typedef __attribute__((ext_vector_type(16))) _Float16 v16h;
typedef __attribute__((ext_vector_type(8)))  float    v8f;

struct cplx { float x, y; };
__device__ __forceinline__ cplx cmul(cplx a, cplx b){ return {a.x*b.x - a.y*b.y, a.x*b.y + a.y*b.x}; }
__device__ __forceinline__ cplx cadd(cplx a, cplx b){ return {a.x+b.x, a.y+b.y}; }
__device__ __forceinline__ cplx csub(cplx a, cplx b){ return {a.x-b.x, a.y-b.y}; }
__device__ __forceinline__ cplx cdiv(cplx a, cplx b){
  float inv = 1.0f/(b.x*b.x + b.y*b.y);
  return {(a.x*b.x + a.y*b.y)*inv, (a.y*b.x - a.x*b.y)*inv};
}
__device__ __forceinline__ cplx conjc(cplx a){ return {a.x, -a.y}; }

// ------------------------- transposes (tiled) ------------------------------
__global__ void __launch_bounds__(256) transpose_u_kernel(const float* __restrict__ u,
                                                          float* __restrict__ ut){
  __shared__ float tile[32][33];
  int b  = blockIdx.z;
  int t0 = blockIdx.x * 32, h0 = blockIdx.y * 32;
  int tx = threadIdx.x, ty = threadIdx.y;
  #pragma unroll
  for (int i = 0; i < 32; i += 8)
    tile[ty + i][tx] = u[((size_t)(b * Lc + t0 + ty + i)) * Hc + h0 + tx];
  __syncthreads();
  #pragma unroll
  for (int i = 0; i < 32; i += 8)
    ut[((size_t)(b * Hc + h0 + ty + i)) * Lc + t0 + tx] = tile[tx][ty + i];
}

__global__ void __launch_bounds__(256) transpose_y_kernel(const float* __restrict__ yt,
                                                          float* __restrict__ y){
  __shared__ float tile[32][33];
  int b  = blockIdx.z;
  int t0 = blockIdx.x * 32, h0 = blockIdx.y * 32;
  int tx = threadIdx.x, ty = threadIdx.y;
  #pragma unroll
  for (int i = 0; i < 32; i += 8)
    tile[ty + i][tx] = yt[((size_t)(b * Hc + h0 + ty + i)) * Lc + t0 + tx];
  __syncthreads();
  #pragma unroll
  for (int i = 0; i < 32; i += 8)
    y[((size_t)(b * Lc + t0 + ty + i)) * Hc + h0 + tx] = tile[tx][ty + i];
}

// ---------------- Abar via Sherman-Morrison (diag + rank-1) ----------------
// I-(dt/2)A = Dm + c P Q^H ; Abar = diag(Dp/Dm) - (P/Dm) z^H (closed form).
__global__ void __launch_bounds__(64) build_abar_kernel(
    const float* __restrict__ lam_re, const float* __restrict__ lam_im,
    const float* __restrict__ p_re,   const float* __restrict__ p_im,
    const float* __restrict__ q_re,   const float* __restrict__ q_im,
    const float* __restrict__ delta,  float* __restrict__ abar){
  int h = blockIdx.x, n = threadIdx.x;
  __shared__ float zr[Nc], zi[Nc], redr[Nc], redi[Nc], sS[2];
  float c = 0.5f * delta[0];
  int idx = h * Nc + n;
  cplx lam = {lam_re[idx], lam_im[idx]};
  cplx P   = {p_re[idx],   p_im[idx]};
  cplx Q   = {q_re[idx],   q_im[idx]};
  cplx Dm  = {1.0f - c*lam.x, -c*lam.y};
  cplx Dp  = {1.0f + c*lam.x,  c*lam.y};
  cplx Dd  = cdiv(Dp, Dm);
  cplx un  = cdiv(P, Dm);            // Dm^{-1} P
  cplx trm = cmul(conjc(Q), un);     // conj(Q)_n P_n / Dm_n
  redr[n] = trm.x; redi[n] = trm.y;
  __syncthreads();
  if (n == 0){
    float sr = 0.f, si = 0.f;
    for (int i = 0; i < Nc; ++i){ sr += redr[i]; si += redi[i]; }
    sS[0] = sr; sS[1] = si;
  }
  __syncthreads();
  cplx s     = {sS[0], sS[1]};
  cplx denom = {1.0f + c*s.x, c*s.y};
  cplx a1    = cdiv({c, 0.f}, denom);                     // c/denom
  cplx a0    = csub({c, 0.f}, cmul({c, 0.f}, cmul(s, a1))); // c - c^2 s/denom
  cplx zh    = cmul(conjc(Q), cadd(a0, cmul(a1, Dd)));    // z^H_m (m := n here)
  zr[n] = zh.x; zi[n] = zh.y;
  __syncthreads();
  float* ar = abar + (size_t)h * (Nc*Nc*2);
  float* ai = ar + Nc*Nc;
  for (int m = 0; m < Nc; ++m){
    float rr = -(un.x * zr[m] - un.y * zi[m]);
    float ii = -(un.x * zi[m] + un.y * zr[m]);
    if (m == n){ rr += Dd.x; ii += Dd.y; }
    ar[n*Nc + m] = rr;  ai[n*Nc + m] = ii;
  }
}

// --------------------- WMMA fragment helpers (ISA 7.12.2) ------------------
__device__ __forceinline__ v16h load_a_frag(const _Float16* M, int row0, int k0, int lane){
  v16h a;
  int m  = row0 + (lane & 15);
  int kb = k0 + ((lane & 16) ? 8 : 0);
  const _Float16* p = M + m * Nc;
  #pragma unroll
  for (int j = 0; j < 8; ++j) a[j]     = p[kb + j];
  #pragma unroll
  for (int j = 0; j < 8; ++j) a[8 + j] = p[kb + 16 + j];
  return a;
}
__device__ __forceinline__ v16h load_b_frag(const _Float16* M, int k0, int col0, int lane){
  v16h b;
  int n  = col0 + (lane & 15);
  int kb = k0 + ((lane & 16) ? 8 : 0);
  #pragma unroll
  for (int j = 0; j < 16; ++j) b[j] = M[(kb + j + (j & 8)) * Nc + n];
  return b;
}

// --------------- complex 64x64 matrix squaring via WMMA --------------------
// out = in*in (complex).  re = Xr*Xr + Xi*(-Xi) ; im = Xr*Xi + Xi*Xr
__global__ void __launch_bounds__(256) matsq_kernel(const float* __restrict__ in,
                                                    float* __restrict__ out){
  __shared__ _Float16 Xre[Nc*Nc];
  __shared__ _Float16 Xim[Nc*Nc];
  int h = blockIdx.x, tid = threadIdx.x;
  const float* inr = in + (size_t)h * (Nc*Nc*2);
  const float* ini = inr + Nc*Nc;
  for (int i = tid; i < Nc*Nc; i += 256){
    Xre[i] = (_Float16)inr[i];
    Xim[i] = (_Float16)ini[i];
  }
  __syncthreads();
  int wave = tid >> 5, lane = tid & 31;
  float* outr = out + (size_t)h * (Nc*Nc*2);
  float* outi = outr + Nc*Nc;
  for (int t = wave; t < 16; t += 8){
    int i0 = (t >> 2) * 16, j0 = (t & 3) * 16;
    v16h ar0 = load_a_frag(Xre, i0, 0,  lane);
    v16h ar1 = load_a_frag(Xre, i0, 32, lane);
    v16h ai0 = load_a_frag(Xim, i0, 0,  lane);
    v16h ai1 = load_a_frag(Xim, i0, 32, lane);
    v16h br0 = load_b_frag(Xre, 0,  j0, lane);
    v16h br1 = load_b_frag(Xre, 32, j0, lane);
    v16h bi0 = load_b_frag(Xim, 0,  j0, lane);
    v16h bi1 = load_b_frag(Xim, 32, j0, lane);
    v16h nbi0, nbi1;
    #pragma unroll
    for (int j = 0; j < 16; ++j){ nbi0[j] = -bi0[j]; nbi1[j] = -bi1[j]; }
    v8f cre = {};
    cre = __builtin_amdgcn_wmma_f32_16x16x32_f16(false, ar0, false, br0,  (short)0, cre, false, false);
    cre = __builtin_amdgcn_wmma_f32_16x16x32_f16(false, ar1, false, br1,  (short)0, cre, false, false);
    cre = __builtin_amdgcn_wmma_f32_16x16x32_f16(false, ai0, false, nbi0, (short)0, cre, false, false);
    cre = __builtin_amdgcn_wmma_f32_16x16x32_f16(false, ai1, false, nbi1, (short)0, cre, false, false);
    v8f cim = {};
    cim = __builtin_amdgcn_wmma_f32_16x16x32_f16(false, ar0, false, bi0,  (short)0, cim, false, false);
    cim = __builtin_amdgcn_wmma_f32_16x16x32_f16(false, ar1, false, bi1,  (short)0, cim, false, false);
    cim = __builtin_amdgcn_wmma_f32_16x16x32_f16(false, ai0, false, br0,  (short)0, cim, false, false);
    cim = __builtin_amdgcn_wmma_f32_16x16x32_f16(false, ai1, false, br1,  (short)0, cim, false, false);
    int col = j0 + (lane & 15);
    int rb  = i0 + ((lane & 16) ? 8 : 0);
    #pragma unroll
    for (int k = 0; k < 8; ++k){
      outr[(rb + k) * Nc + col] = cre[k];
      outi[(rb + k) * Nc + col] = cim[k];
    }
  }
}

// ---- Ct = conj(I - Abar^L)^T C ; Cauchy weights w00..w11 per channel ------
__global__ void __launch_bounds__(64) ct_weights_kernel(
    const float* __restrict__ abarL,
    const float* __restrict__ p_re, const float* __restrict__ p_im,
    const float* __restrict__ q_re, const float* __restrict__ q_im,
    const float* __restrict__ b_re, const float* __restrict__ b_im,
    const float* __restrict__ c_re, const float* __restrict__ c_im,
    float* __restrict__ wts){
  int h = blockIdx.x, m = threadIdx.x;
  const float* alr = abarL + (size_t)h * (Nc*Nc*2);
  const float* ali = alr + Nc*Nc;
  cplx ct = {0.f, 0.f};
  for (int n = 0; n < Nc; ++n){
    float zr = ((n == m) ? 1.0f : 0.0f) - alr[n*Nc + m];   // (I - AbarL)[n][m]
    float zi = -ali[n*Nc + m];
    cplx Cn = {c_re[h*Nc + n], c_im[h*Nc + n]};
    ct.x += zr*Cn.x + zi*Cn.y;                              // conj(z)*C
    ct.y += zr*Cn.y - zi*Cn.x;
  }
  int idx = h*Nc + m;
  cplx P  = {p_re[idx], p_im[idx]};
  cplx Q  = {q_re[idx], q_im[idx]};
  cplx Bv = {b_re[idx], b_im[idx]};
  cplx cc = conjc(ct), cq = conjc(Q);
  cplx w00 = cmul(cc, Bv), w01 = cmul(cc, P);
  cplx w10 = cmul(cq, Bv), w11 = cmul(cq, P);
  float* w = wts + (size_t)h * (8*Nc);
  w[0*Nc+m] = w00.x; w[1*Nc+m] = w00.y;
  w[2*Nc+m] = w01.x; w[3*Nc+m] = w01.y;
  w[4*Nc+m] = w10.x; w[5*Nc+m] = w10.y;
  w[6*Nc+m] = w11.x; w[7*Nc+m] = w11.y;
}

// ------------- Cauchy kernel + Woodbury -> Khat (H,L) complex --------------
__global__ void __launch_bounds__(256) cauchy_kernel(
    const float* __restrict__ lam_re, const float* __restrict__ lam_im,
    const float* __restrict__ wts, const float* __restrict__ delta,
    float* __restrict__ khat){
  __shared__ float lamS[2*Nc];
  __shared__ float wS[8*Nc];
  int h = blockIdx.y, tid = threadIdx.x;
  for (int i = tid; i < 8*Nc; i += 256) wS[i] = wts[(size_t)h*(8*Nc) + i];
  if (tid < Nc){
    lamS[tid]      = lam_re[h*Nc + tid];
    lamS[Nc + tid] = lam_im[h*Nc + tid];
  }
  __syncthreads();
  int l = blockIdx.x * 256 + tid;
  float dt = delta[0];
  float ang = 6.2831853071795864769f * ((float)l / (float)Lc);
  float sn, cs; sincosf(ang, &sn, &cs);
  cplx om = {cs, sn};
  cplx opo = {1.0f + om.x, om.y};
  cplx g = cmul({2.0f / dt, 0.f}, cdiv({1.0f - om.x, -om.y}, opo));
  cplx k00 = {0,0}, k01 = {0,0}, k10 = {0,0}, k11 = {0,0};
  #pragma unroll 4
  for (int n = 0; n < Nc; ++n){
    float dxr = g.x - lamS[n], dxi = g.y - lamS[Nc + n];
    float inv = 1.0f / (dxr*dxr + dxi*dxi);
    cplx r = {dxr*inv, -dxi*inv};                    // 1/(g-lam)
    cplx w;
    w = {wS[0*Nc+n], wS[1*Nc+n]}; k00 = cadd(k00, cmul(r, w));
    w = {wS[2*Nc+n], wS[3*Nc+n]}; k01 = cadd(k01, cmul(r, w));
    w = {wS[4*Nc+n], wS[5*Nc+n]}; k10 = cadd(k10, cmul(r, w));
    w = {wS[6*Nc+n], wS[7*Nc+n]}; k11 = cadd(k11, cmul(r, w));
  }
  cplx kh   = csub(k00, cdiv(cmul(k01, k10), {1.0f + k11.x, k11.y}));
  cplx pref = cdiv({2.0f, 0.f}, opo);
  kh = cmul(pref, kh);
  khat[(size_t)h * (2*Lc) + l]      = kh.x;
  khat[(size_t)h * (2*Lc) + Lc + l] = kh.y;
}

// ---------------------- in-LDS radix-2 complex FFT -------------------------
__device__ void block_fft(float* re, float* im, int n, int log2n, float sign){
  int tid = threadIdx.x, nt = blockDim.x;
  for (int i = tid; i < n; i += nt){
    int j = (int)(__brev((unsigned)i) >> (32 - log2n));
    if (j > i){
      float tr = re[i]; re[i] = re[j]; re[j] = tr;
      float ti = im[i]; im[i] = im[j]; im[j] = ti;
    }
  }
  __syncthreads();
  for (int s = 1; s <= log2n; ++s){
    int mh = 1 << (s - 1);
    for (int k = tid; k < (n >> 1); k += nt){
      int j  = k & (mh - 1);
      int i1 = ((k >> (s - 1)) << s) + j;
      int i2 = i1 + mh;
      float ang = sign * 3.14159265358979f * (float)j / (float)mh;  // 2*pi*j/m
      float sn, cs; sincosf(ang, &sn, &cs);
      float xr = re[i2], xi = im[i2];
      float tr = cs*xr - sn*xi, ti = cs*xi + sn*xr;
      float ur = re[i1], ui = im[i1];
      re[i1] = ur + tr; im[i1] = ui + ti;
      re[i2] = ur - tr; im[i2] = ui - ti;
    }
    __syncthreads();
  }
}

// ---- K = ifft4096(Khat).real ; kf = fft8192(pad(K)) per channel -----------
__global__ void __launch_bounds__(256) kbuild_kernel(const float* __restrict__ khat,
                                                     float* __restrict__ kf){
  extern __shared__ float sm[];
  float* re = sm;
  float* im = sm + L2;
  int h = blockIdx.x, tid = threadIdx.x;
  const float* kr = khat + (size_t)h * (2*Lc);
  for (int i = tid; i < Lc; i += 256){ re[i] = kr[i]; im[i] = kr[Lc + i]; }
  __syncthreads();
  block_fft(re, im, Lc, 12, 1.0f);                    // inverse FFT (unscaled)
  for (int i = tid; i < Lc; i += 256){
    re[i] *= (1.0f / (float)Lc);                      // K[t] = Re(ifft)/L
    im[i] = 0.0f;
    re[Lc + i] = 0.0f; im[Lc + i] = 0.0f;             // zero-pad to 2L
  }
  __syncthreads();
  block_fft(re, im, L2, 13, -1.0f);                   // forward FFT
  float* ko = kf + (size_t)h * (2*L2);
  for (int i = tid; i < L2; i += 256){ ko[i] = re[i]; ko[L2 + i] = im[i]; }
}

// -------- per-(b,h) row: y = ifft( fft(pad(u)) * kf )[:L] + d*u ------------
__global__ void __launch_bounds__(256) conv_kernel(float* __restrict__ ut,
                                                   const float* __restrict__ kf,
                                                   const float* __restrict__ d){
  extern __shared__ float sm[];
  float* re = sm;
  float* im = sm + L2;
  int b = blockIdx.x >> 8, h = blockIdx.x & 255;
  int tid = threadIdx.x;
  float* urow = ut + ((size_t)b * Hc + h) * Lc;
  for (int i = tid; i < L2; i += 256){
    re[i] = (i < Lc) ? urow[i] : 0.0f;
    im[i] = 0.0f;
  }
  __syncthreads();
  block_fft(re, im, L2, 13, -1.0f);
  const float* kfr = kf + (size_t)h * (2*L2);
  const float* kfi = kfr + L2;
  for (int k = tid; k < L2; k += 256){
    float xr = re[k], xi = im[k];
    float wr = kfr[k], wi = kfi[k];
    re[k] = xr*wr - xi*wi;
    im[k] = xr*wi + xi*wr;
  }
  __syncthreads();
  block_fft(re, im, L2, 13, 1.0f);
  float dh = d[h];
  const float inv = 1.0f / (float)L2;
  for (int t = tid; t < Lc; t += 256){
    float uv = urow[t];
    urow[t] = re[t] * inv + dh * uv;                  // in-place: row fully read above
  }
}

// ---------------------------------------------------------------------------
extern "C" void kernel_launch(void* const* d_in, const int* in_sizes, int n_in,
                              void* d_out, int out_size, void* d_ws, size_t ws_size,
                              hipStream_t stream){
  (void)in_sizes; (void)n_in; (void)out_size; (void)ws_size;
  const float* lam_re = (const float*)d_in[0];
  const float* lam_im = (const float*)d_in[1];
  const float* p_re   = (const float*)d_in[2];
  const float* p_im   = (const float*)d_in[3];
  const float* q_re   = (const float*)d_in[4];
  const float* q_im   = (const float*)d_in[5];
  const float* b_re   = (const float*)d_in[6];
  const float* b_im   = (const float*)d_in[7];
  const float* c_re   = (const float*)d_in[8];
  const float* c_im   = (const float*)d_in[9];
  const float* dvec   = (const float*)d_in[10];
  const float* delta  = (const float*)d_in[11];
  const float* u      = (const float*)d_in[12];
  float* y  = (float*)d_out;
  float* ws = (float*)d_ws;

  // workspace layout (floats)
  const size_t abar_sz = (size_t)Hc * Nc * Nc * 2;     // 2,097,152
  float* abarA = ws;
  float* abarB = abarA + abar_sz;
  float* wts   = abarB + abar_sz;                      // H*8*64 = 131072
  float* khat  = wts + (size_t)Hc * 8 * Nc;            // H*2*L  = 2,097,152
  float* kf    = khat + (size_t)Hc * 2 * Lc;           // H*2*2L = 4,194,304
  float* ut    = kf + (size_t)Hc * 2 * L2;             // B*H*L  = 8,388,608

  // 1. transpose u (B,L,H) -> ut (B,H,L)
  transpose_u_kernel<<<dim3(Lc/32, Hc/32, Bc), dim3(32, 8), 0, stream>>>(u, ut);

  // 2. Abar via Sherman-Morrison closed form
  build_abar_kernel<<<Hc, Nc, 0, stream>>>(lam_re, lam_im, p_re, p_im,
                                           q_re, q_im, delta, abarA);

  // 3. Abar^4096 = 12 repeated complex squarings (WMMA f16->f32)
  float* cur = abarA; float* nxt = abarB;
  for (int it = 0; it < 12; ++it){
    matsq_kernel<<<Hc, 256, 0, stream>>>(cur, nxt);
    float* t = cur; cur = nxt; nxt = t;
  }
  // after 12 swaps, result is back in abarA (== cur)

  // 4. Ct and Cauchy weights
  ct_weights_kernel<<<Hc, Nc, 0, stream>>>(cur, p_re, p_im, q_re, q_im,
                                           b_re, b_im, c_re, c_im, wts);

  // 5. Cauchy kernel + Woodbury -> Khat
  cauchy_kernel<<<dim3(Lc/256, Hc), 256, 0, stream>>>(lam_re, lam_im, wts, delta, khat);

  // 6. K = ifft(Khat).real ; kf = fft8192(pad(K))
  kbuild_kernel<<<Hc, 256, 2 * L2 * sizeof(float), stream>>>(khat, kf);

  // 7. FFT convolution per (b,h) row, + d*u skip (in place in ut)
  conv_kernel<<<Bc * Hc, 256, 2 * L2 * sizeof(float), stream>>>(ut, kf, dvec);

  // 8. transpose back (B,H,L) -> (B,L,H)
  transpose_y_kernel<<<dim3(Lc/32, Hc/32, Bc), dim3(32, 8), 0, stream>>>(ut, y);
}
